// Nucleus1SpatialAxialAttention_1829656068445
// MI455X (gfx1250) — matrix operand order repaired and checked by reading, not compiled, not run on hardware
//
#include <hip/hip_runtime.h>
#include <hip/hip_bf16.h>

// ---------------------------------------------------------------------------
// Nucleus1 spatial axial attention, CDNA5 (gfx1250), wave32 + WMMA bf16.
// - All WMMA operands staged in LDS in fragment-friendly layouts (b128 loads)
// - Q/K attention staging + out-GEMM A staging use async global->LDS copies
//   (GLOBAL_LOAD_ASYNC_TO_LDS_B128, ASYNCcnt-tracked)
// ---------------------------------------------------------------------------

typedef __attribute__((ext_vector_type(16))) __bf16        v16bf;
typedef __attribute__((ext_vector_type(8)))  __bf16        v8bf;
typedef __attribute__((ext_vector_type(8)))  float         v8f;
typedef __attribute__((ext_vector_type(4)))  float         f32x4;
typedef __attribute__((ext_vector_type(4)))  unsigned int  u32x4;
typedef __attribute__((ext_vector_type(8)))  unsigned short u16x8;

#define N_TOK   65536      // 16 * 64 * 64
#define CDIM    768
#define C3      2304
#define HEADS   8
#define HDIM    96
#define LINES   64
#define PI_F    3.14159265358979f

// ---- bf16 <-> f32 helpers -------------------------------------------------
__device__ __forceinline__ unsigned short f2b(float f) {
    return __builtin_bit_cast(unsigned short, (__bf16)f);   // native cvt (RNE)
}
__device__ __forceinline__ float b2f(unsigned short h) {
    return __builtin_bit_cast(float, (unsigned)h << 16);
}

// ---- async global -> LDS copy (16B per lane), ASYNCcnt-tracked ------------
__device__ __forceinline__ void async_cp16(void* lds_dst, const void* gsrc) {
    unsigned       ldsa = (unsigned)(unsigned long long)lds_dst;  // LDS byte addr
    unsigned long long ga = (unsigned long long)gsrc;
    asm volatile("global_load_async_to_lds_b128 %0, %1, off"
                 :: "v"(ldsa), "v"(ga) : "memory");
}
__device__ __forceinline__ void async_wait0() {
    asm volatile("s_wait_asynccnt 0x0" ::: "memory");
}

// ---- WMMA fragment loaders (wave32 layouts per CDNA5 ISA 7.12.2) ----------
// A fragment: rows m0..m0+15, cols k0..k0+31 of row-major bf16 matrix.
// lane<16: row=m0+lane, K={0..7,16..23}; lane>=16: same row, K={8..15,24..31}.
// Requires ld % 8 == 0 and k0 % 8 == 0 so both 8-element runs are 16B aligned.
__device__ __forceinline__ v16bf load_a_frag(const unsigned short* A, int ld,
                                             int m0, int k0) {
    const int l = threadIdx.x & 31;
    const unsigned short* p = A + (m0 + (l & 15)) * ld + k0 + (l >> 4) * 8;
    v8bf lo = __builtin_bit_cast(v8bf, *(const u32x4*)p);          // K + 0..7
    v8bf hi = __builtin_bit_cast(v8bf, *(const u32x4*)(p + 16));   // K + 16..23
    return __builtin_shufflevector(lo, hi, 0, 1, 2, 3, 4, 5, 6, 7,
                                           8, 9, 10, 11, 12, 13, 14, 15);
}
// B fragment of M^T: B_tile(kk,n) = M[n0+n][k0+kk], M row-major stride ld.
__device__ __forceinline__ v16bf load_bt_frag(const unsigned short* M, int ld,
                                              int k0, int n0) {
    const int l = threadIdx.x & 31;
    const unsigned short* p = M + (n0 + (l & 15)) * ld + k0 + (l >> 4) * 8;
    v8bf lo = __builtin_bit_cast(v8bf, *(const u32x4*)p);
    v8bf hi = __builtin_bit_cast(v8bf, *(const u32x4*)(p + 16));
    return __builtin_shufflevector(lo, hi, 0, 1, 2, 3, 4, 5, 6, 7,
                                           8, 9, 10, 11, 12, 13, 14, 15);
}

// ---------------------------------------------------------------------------
// Kernel 1: qkv = bf16(x) @ w_in + b_in      (65536 x 768) * (768 x 2304)
// block = 256 (8 waves), block tile 64x128, wave tile 32x32 (2x2 WMMA tiles)
// ---------------------------------------------------------------------------
__global__ void __launch_bounds__(256)
qkv_gemm_kernel(const float* __restrict__ x,
                const unsigned short* __restrict__ w_in,
                const unsigned short* __restrict__ b_in,
                unsigned short* __restrict__ qkv) {
    const int gm0 = blockIdx.y * 64;
    const int gn0 = blockIdx.x * 128;
    const int wave = threadIdx.x >> 5;
    const int wm0 = (wave >> 2) * 32;            // 0 or 32
    const int wn0 = (wave & 3) * 32;             // 0, 32, 64, 96

    __shared__ alignas(16) unsigned short ldsA[64][40];    // 64x32 (+pad)
    __shared__ alignas(16) unsigned short ldsBt[128][32];  // B^T: [n][k]

    const int ar  = threadIdx.x >> 2;            // A: row 0..63
    const int ac0 = (threadIdx.x & 3) * 8;       // A: col 0,8,16,24
    const int br  = threadIdx.x >> 3;            // B: k row 0..31
    const int bc0 = (threadIdx.x & 7) * 16;      // B: n col 0..112 step 16

    v8f acc[2][2] = {};
    for (int k0 = 0; k0 < CDIM; k0 += 32) {
        // ---- stage A tile (fp32 -> bf16 via native cvt) -------------------
        const float* xp = x + (size_t)(gm0 + ar) * CDIM + k0 + ac0;
        f32x4 a0 = *(const f32x4*)xp;
        f32x4 a1 = *(const f32x4*)(xp + 4);
        u16x8 pk;
#pragma unroll
        for (int e = 0; e < 4; ++e) { pk[e] = f2b(a0[e]); pk[e + 4] = f2b(a1[e]); }
        *(u32x4*)&ldsA[ar][ac0] = __builtin_bit_cast(u32x4, pk);

        // ---- stage B tile transposed --------------------------------------
        const unsigned short* wp = w_in + (size_t)(k0 + br) * C3 + gn0 + bc0;
        u16x8 b0 = __builtin_bit_cast(u16x8, *(const u32x4*)wp);
        u16x8 b1 = __builtin_bit_cast(u16x8, *(const u32x4*)(wp + 8));
#pragma unroll
        for (int e = 0; e < 8; ++e) {
            ldsBt[bc0 + e][br]     = b0[e];
            ldsBt[bc0 + 8 + e][br] = b1[e];
        }
        if (k0 + 32 < CDIM) {
            __builtin_prefetch(xp + 32, 0, 0);
            __builtin_prefetch(wp + (size_t)32 * C3, 0, 0);
        }
        __syncthreads();

        v16bf af0 = load_a_frag(&ldsA[0][0], 40, wm0, 0);
        v16bf af1 = load_a_frag(&ldsA[0][0], 40, wm0 + 16, 0);
        v16bf bf0 = load_bt_frag(&ldsBt[0][0], 32, 0, wn0);
        v16bf bf1 = load_bt_frag(&ldsBt[0][0], 32, 0, wn0 + 16);
        acc[0][0] = __builtin_amdgcn_wmma_f32_16x16x32_bf16(
                        false, af0, false, bf0, (short)0, acc[0][0], false, false);
        acc[0][1] = __builtin_amdgcn_wmma_f32_16x16x32_bf16(
                        false, af0, false, bf1, (short)0, acc[0][1], false, false);
        acc[1][0] = __builtin_amdgcn_wmma_f32_16x16x32_bf16(
                        false, af1, false, bf0, (short)0, acc[1][0], false, false);
        acc[1][1] = __builtin_amdgcn_wmma_f32_16x16x32_bf16(
                        false, af1, false, bf1, (short)0, acc[1][1], false, false);
        __syncthreads();
    }

    const int l = threadIdx.x & 31;
    const int half = l >> 4;
#pragma unroll
    for (int nj = 0; nj < 2; ++nj) {
        const int col = gn0 + wn0 + nj * 16 + (l & 15);
        const float bias = b2f(b_in[col]);
#pragma unroll
        for (int mi = 0; mi < 2; ++mi)
#pragma unroll
            for (int r = 0; r < 8; ++r) {
                int row = gm0 + wm0 + mi * 16 + r + 8 * half;
                qkv[(size_t)row * C3 + col] = f2b(acc[mi][nj][r] + bias);
            }
    }
}

// ---------------------------------------------------------------------------
// Kernel 2: fused per-head LayerNorm(96) + affine + axial RoPE, in place on
// the q and k thirds of qkv. One thread per (token, head, q|k).
// ---------------------------------------------------------------------------
__global__ void __launch_bounds__(256)
ln_rope_kernel(unsigned short* __restrict__ qkv,
               const unsigned short* __restrict__ q_gamma,
               const unsigned short* __restrict__ q_beta,
               const unsigned short* __restrict__ k_gamma,
               const unsigned short* __restrict__ k_beta) {
    const int idx   = blockIdx.x * blockDim.x + threadIdx.x;
    const int which = idx & 1;          // 0 = q, 1 = k
    const int head  = (idx >> 1) & 7;
    const int n     = idx >> 4;         // token
    if (n >= N_TOK) return;

    unsigned* v32 = (unsigned*)(qkv + (size_t)n * C3 + which * CDIM + head * HDIM);
    const unsigned* g32 = (const unsigned*)(which ? k_gamma : q_gamma);
    const unsigned* b32 = (const unsigned*)(which ? k_beta  : q_beta);

    float s = 0.f, ss = 0.f;
#pragma unroll
    for (int i = 0; i < HDIM / 2; ++i) {
        unsigned u = v32[i];
        float a = b2f((unsigned short)u);
        float b = b2f((unsigned short)(u >> 16));
        s += a + b; ss += a * a + b * b;
    }
    const float mu  = s * (1.f / HDIM);
    const float var = ss * (1.f / HDIM) - mu * mu;
    const float inv = rsqrtf(var + 1e-5f);

    const int hh = (n >> 6) & 63;
    const int ww = n & 63;
    const float posh = -1.f + 2.f * hh * (1.f / 63.f);
    const float posw = -1.f + 2.f * ww * (1.f / 63.f);

#pragma unroll
    for (int i = 0; i < HDIM / 2; ++i) {
        unsigned u  = v32[i];
        unsigned ug = g32[i];
        unsigned ub = b32[i];
        float e = (b2f((unsigned short)u)         - mu) * inv
                    * b2f((unsigned short)ug)         + b2f((unsigned short)ub);
        float o = (b2f((unsigned short)(u >> 16)) - mu) * inv
                    * b2f((unsigned short)(ug >> 16)) + b2f((unsigned short)(ub >> 16));
        if (i < 32) {
            float base = (1.f + 127.f * (float)(i & 15) * (1.f / 15.f)) * PI_F;
            float f = (i < 16 ? posh : posw) * base;
            float c = cosf(f), sn = sinf(f);
            float e2 = e * c - o * sn;
            o = o * c + e * sn;
            e = e2;
        }
        v32[i] = (unsigned)f2b(e) | ((unsigned)f2b(o) << 16);
    }
}

// ---------------------------------------------------------------------------
// Kernel 3: one workgroup per (t, head, line): SDPA over a 64-token line.
// mode 0 (axis-x, stride 1): o  = softmax(QK^T * s) V
// mode 1 (axis-y, stride 64): o = 0.5 * (o_prev + softmax(QK^T * s) V)
// block = 128 (4 waves; wave w owns S/O rows 16w..16w+15); grid = 8192
// ---------------------------------------------------------------------------
__global__ void __launch_bounds__(128)
attn_line_kernel(const unsigned short* __restrict__ qkv,
                 unsigned short* __restrict__ o,
                 int mode) {
    const int line = blockIdx.x & 63;
    const int head = (blockIdx.x >> 6) & 7;
    const int t    = blockIdx.x >> 9;

    const size_t n_base   = (size_t)t * 4096 + (mode == 0 ? line * 64 : line);
    const int    n_stride = (mode == 0 ? 1 : 64);

    __shared__ alignas(16) unsigned short Qs[64][HDIM];
    __shared__ alignas(16) unsigned short Ks[64][HDIM];
    __shared__ alignas(16) unsigned short Vt[HDIM][64];   // V transposed: [d][token]
    __shared__ alignas(16) float          Ss[64][65];
    __shared__ alignas(16) unsigned short Ps[64][64];
    __shared__ float                      red[128];

    // ---- stage: Q/K via async global->LDS (b128), V transposed manually ---
    {
        const int r     = threadIdx.x >> 1;            // token row 0..63
        const int cbase = (threadIdx.x & 1) * 48;      // half-row of 48 elems
        const unsigned short* src =
            qkv + (n_base + (size_t)r * n_stride) * C3 + head * HDIM;
#pragma unroll
        for (int j = 0; j < 6; ++j) {
            const int c = cbase + j * 8;
            async_cp16(&Qs[r][c], src + c);            // ASYNCcnt-tracked DMA
            async_cp16(&Ks[r][c], src + CDIM + c);
            u16x8 vv = __builtin_bit_cast(u16x8, *(const u32x4*)(src + 2 * CDIM + c));
#pragma unroll
            for (int e = 0; e < 8; ++e) Vt[c + e][r] = vv[e];
        }
    }
    async_wait0();
    __syncthreads();

    const int wave = threadIdx.x >> 5;
    const int m0   = wave * 16;
    const int l    = threadIdx.x & 31;
    const int half = l >> 4;

    // ---- S = Q K^T * scale (4 tiles of 16x16 per wave, K = 96 = 3*32) ----
    v8f sacc[4] = {};
    for (int kb = 0; kb < HDIM; kb += 32) {
        v16bf aq = load_a_frag(&Qs[0][0], HDIM, m0, kb);
#pragma unroll
        for (int j = 0; j < 4; ++j) {
            v16bf bk = load_bt_frag(&Ks[0][0], HDIM, kb, j * 16);
            sacc[j] = __builtin_amdgcn_wmma_f32_16x16x32_bf16(
                        false, aq, false, bk, (short)0, sacc[j], false, false);
        }
    }
    const float scale = 0.102062072615966f;   // 1/sqrt(96)
#pragma unroll
    for (int j = 0; j < 4; ++j)
#pragma unroll
        for (int r = 0; r < 8; ++r)
            Ss[m0 + r + 8 * half][j * 16 + (l & 15)] = sacc[j][r] * scale;
    __syncthreads();

    // ---- fp32 row softmax (2 threads per row), P -> bf16 in LDS ----------
    const int row  = threadIdx.x >> 1;
    const int part = threadIdx.x & 1;
    float mx = -3.4e38f;
    for (int c = part * 32; c < part * 32 + 32; ++c) mx = fmaxf(mx, Ss[row][c]);
    red[threadIdx.x] = mx;
    __syncthreads();
    mx = fmaxf(red[row * 2], red[row * 2 + 1]);
    float sum = 0.f;
    for (int c = part * 32; c < part * 32 + 32; ++c) {
        float e = __expf(Ss[row][c] - mx);
        Ss[row][c] = e;
        sum += e;
    }
    __syncthreads();
    red[threadIdx.x] = sum;
    __syncthreads();
    const float rinv = 1.f / (red[row * 2] + red[row * 2 + 1]);
    for (int c = part * 32; c < part * 32 + 32; c += 2) {
        unsigned u = (unsigned)f2b(Ss[row][c] * rinv)
                   | ((unsigned)f2b(Ss[row][c + 1] * rinv) << 16);
        *(unsigned*)&Ps[row][c] = u;
    }
    __syncthreads();

    // ---- O = P V (6 tiles of 16x16 per wave, K = 64 = 2*32) --------------
    v8f oacc[6] = {};
    for (int kb = 0; kb < 64; kb += 32) {
        v16bf ap = load_a_frag(&Ps[0][0], 64, m0, kb);
#pragma unroll
        for (int j = 0; j < 6; ++j) {
            v16bf bv = load_bt_frag(&Vt[0][0], 64, kb, j * 16);
            oacc[j] = __builtin_amdgcn_wmma_f32_16x16x32_bf16(
                        false, ap, false, bv, (short)0, oacc[j], false, false);
        }
    }

    // ---- epilogue: write xx (mode 0) or 0.5*(xx + xy) (mode 1) -----------
#pragma unroll
    for (int j = 0; j < 6; ++j) {
        const int d = j * 16 + (l & 15);
#pragma unroll
        for (int r = 0; r < 8; ++r) {
            const int m = m0 + r + 8 * half;
            unsigned short* dst =
                o + (n_base + (size_t)m * n_stride) * CDIM + head * HDIM + d;
            float val = oacc[j][r];
            if (mode) val = 0.5f * (b2f(*dst) + val);
            *dst = f2b(val);
        }
    }
}

// ---------------------------------------------------------------------------
// Kernel 4: out = o @ w_out + b_out   (65536 x 768)*(768 x 768) -> fp32
// same blocking as kernel 1; A tile staged via async global->LDS DMA
// ---------------------------------------------------------------------------
__global__ void __launch_bounds__(256)
out_gemm_kernel(const unsigned short* __restrict__ o,
                const unsigned short* __restrict__ w_out,
                const unsigned short* __restrict__ b_out,
                float* __restrict__ out) {
    const int gm0 = blockIdx.y * 64;
    const int gn0 = blockIdx.x * 128;
    const int wave = threadIdx.x >> 5;
    const int wm0 = (wave >> 2) * 32;
    const int wn0 = (wave & 3) * 32;

    __shared__ alignas(16) unsigned short ldsA[64][40];
    __shared__ alignas(16) unsigned short ldsBt[128][32];

    const int ar  = threadIdx.x >> 2;
    const int ac0 = (threadIdx.x & 3) * 8;
    const int br  = threadIdx.x >> 3;
    const int bc0 = (threadIdx.x & 7) * 16;

    v8f acc[2][2] = {};
    for (int k0 = 0; k0 < CDIM; k0 += 32) {
        // A tile: straight row-major copy -> async global->LDS DMA
        const unsigned short* op = o + (size_t)(gm0 + ar) * CDIM + k0 + ac0;
        async_cp16(&ldsA[ar][ac0], op);

        const unsigned short* wp = w_out + (size_t)(k0 + br) * CDIM + gn0 + bc0;
        u16x8 b0 = __builtin_bit_cast(u16x8, *(const u32x4*)wp);
        u16x8 b1 = __builtin_bit_cast(u16x8, *(const u32x4*)(wp + 8));
#pragma unroll
        for (int e = 0; e < 8; ++e) {
            ldsBt[bc0 + e][br]     = b0[e];
            ldsBt[bc0 + 8 + e][br] = b1[e];
        }
        if (k0 + 32 < CDIM) {
            __builtin_prefetch(op + 32, 0, 0);
            __builtin_prefetch(wp + (size_t)32 * CDIM, 0, 0);
        }
        async_wait0();
        __syncthreads();

        v16bf af0 = load_a_frag(&ldsA[0][0], 40, wm0, 0);
        v16bf af1 = load_a_frag(&ldsA[0][0], 40, wm0 + 16, 0);
        v16bf bf0 = load_bt_frag(&ldsBt[0][0], 32, 0, wn0);
        v16bf bf1 = load_bt_frag(&ldsBt[0][0], 32, 0, wn0 + 16);
        acc[0][0] = __builtin_amdgcn_wmma_f32_16x16x32_bf16(
                        false, af0, false, bf0, (short)0, acc[0][0], false, false);
        acc[0][1] = __builtin_amdgcn_wmma_f32_16x16x32_bf16(
                        false, af0, false, bf1, (short)0, acc[0][1], false, false);
        acc[1][0] = __builtin_amdgcn_wmma_f32_16x16x32_bf16(
                        false, af1, false, bf0, (short)0, acc[1][0], false, false);
        acc[1][1] = __builtin_amdgcn_wmma_f32_16x16x32_bf16(
                        false, af1, false, bf1, (short)0, acc[1][1], false, false);
        __syncthreads();
    }

    const int l = threadIdx.x & 31;
    const int half = l >> 4;
#pragma unroll
    for (int nj = 0; nj < 2; ++nj) {
        const int col = gn0 + wn0 + nj * 16 + (l & 15);
        const float bias = b2f(b_out[col]);
#pragma unroll
        for (int mi = 0; mi < 2; ++mi)
#pragma unroll
            for (int r = 0; r < 8; ++r) {
                int row = gm0 + wm0 + mi * 16 + r + 8 * half;
                out[(size_t)row * CDIM + col] = acc[mi][nj][r] + bias;
            }
    }
}

// ---------------------------------------------------------------------------
extern "C" void kernel_launch(void* const* d_in, const int* in_sizes, int n_in,
                              void* d_out, int out_size, void* d_ws, size_t ws_size,
                              hipStream_t stream) {
    (void)in_sizes; (void)n_in; (void)out_size; (void)ws_size;

    const float*          x       = (const float*)d_in[0];
    const unsigned short* w_in    = (const unsigned short*)d_in[1];
    const unsigned short* b_in    = (const unsigned short*)d_in[2];
    const unsigned short* w_out   = (const unsigned short*)d_in[3];
    const unsigned short* b_out   = (const unsigned short*)d_in[4];
    const unsigned short* q_gamma = (const unsigned short*)d_in[5];
    const unsigned short* q_beta  = (const unsigned short*)d_in[6];
    const unsigned short* k_gamma = (const unsigned short*)d_in[7];
    const unsigned short* k_beta  = (const unsigned short*)d_in[8];
    float* out = (float*)d_out;

    unsigned short* qkv  = (unsigned short*)d_ws;                  // 65536*2304 bf16
    unsigned short* obuf = qkv + (size_t)N_TOK * C3;               // 65536*768  bf16

    // 1) QKV projection
    qkv_gemm_kernel<<<dim3(C3 / 128, N_TOK / 64), 256, 0, stream>>>(x, w_in, b_in, qkv);
    // 2) LN + affine + axial RoPE on q, k (in place)
    ln_rope_kernel<<<(N_TOK * HEADS * 2) / 256, 256, 0, stream>>>(
        qkv, q_gamma, q_beta, k_gamma, k_beta);
    // 3) axial attention: axis-x (over W) then axis-y (over H), 0.5*(xx+xy)
    attn_line_kernel<<<16 * HEADS * LINES, 128, 0, stream>>>(qkv, obuf, 0);
    attn_line_kernel<<<16 * HEADS * LINES, 128, 0, stream>>>(qkv, obuf, 1);
    // 4) output projection (fp32 out)
    out_gemm_kernel<<<dim3(CDIM / 128, N_TOK / 64), 256, 0, stream>>>(obuf, w_out, b_out, out);
}
